// MultiHeadAttention_KT_14912126452241
// MI455X (gfx1250) — compile-verified, matrix operand order checked by reading, not run on hardware
//
#include <hip/hip_runtime.h>

// ---------------------------------------------------------------------------
// MI455X (gfx1250, wave32) masked multi-head *linear* causal attention.
//   - All GEMMs on v_wmma_f32_16x16x32_bf16 (compute-bound: ~34 GFLOP GEMM vs
//     ~85MB traffic -> bf16 matrix path is the roofline winner).
//   - fp32->bf16 conversion + masking hoisted into cheap prep passes so the
//     GEMM tiles are pure bf16 2D tiles => staged into LDS by the Tensor Data
//     Mover (tensor_load_to_lds + s_wait_tensorcnt), double buffered.
//   - Chunked linear-attention recurrence (chunk=64) with the 64x64 fp32
//     state held in WMMA f32 accumulators across the sequence.
// ---------------------------------------------------------------------------

typedef __bf16 bf16_t;
typedef __attribute__((ext_vector_type(16))) __bf16 v16bf;
typedef __attribute__((ext_vector_type(8)))  float  v8f;
typedef __attribute__((ext_vector_type(4)))  unsigned int v4u;
typedef __attribute__((ext_vector_type(8)))  int v8i;
typedef __attribute__((ext_vector_type(4)))  int v4i;

#define WMMA_BF16(A, B, C) \
  __builtin_amdgcn_wmma_f32_16x16x32_bf16(false, (A), false, (B), (short)0, (C), false, false)

#if __has_include(<hip/amd_detail/amd_gfx1250_TDM.h>)
#define TDM_ARGS6 1
#else
#define TDM_ARGS6 0
#endif

__device__ __forceinline__ v8f v8f_zero() {
  v8f z;
#pragma unroll
  for (int i = 0; i < 8; ++i) z[i] = 0.0f;
  return z;
}

// Issue a TDM 2D tile load: bf16 tensor [tensor_d1][tensor_d0] row-major with
// row stride `stride0` (elements); tile [tile_d1][tile_d0] starting at gptr,
// landing at LDS offset lds_off with pad_enable (LDS row stride padded).
// pad codes: pad every 2^(ivl+1) DWORDs by (amt+1) DWORDs.
__device__ __forceinline__ void tdm_load_2d_bf16(const void* gptr, unsigned lds_off,
                                                 unsigned tensor_d0, unsigned tensor_d1,
                                                 unsigned long long stride0,
                                                 unsigned tile_d0, unsigned tile_d1,
                                                 unsigned pad_ivl, unsigned pad_amt) {
  const unsigned long long ga = (unsigned long long)gptr;
  v4u g0;
  g0[0] = 1u;                                                // count=1 (valid D#)
  g0[1] = lds_off;                                           // lds_addr
  g0[2] = (unsigned)(ga & 0xffffffffu);                      // global_addr lo
  g0[3] = (unsigned)(((ga >> 32) & 0x1ffffffu) | (2u << 30)); // addr hi | type=2
  v8i g1;
  g1[0] = (int)((1u << 16) | (1u << 20) | (pad_ivl << 22) | (pad_amt << 25)); // 2B elems, pad_enable
  g1[1] = (int)((tensor_d0 & 0xffffu) << 16);
  g1[2] = (int)(((tensor_d0 >> 16) & 0xffffu) | ((tensor_d1 & 0xffffu) << 16));
  g1[3] = (int)(((tensor_d1 >> 16) & 0xffffu) | ((tile_d0 & 0xffffu) << 16));
  g1[4] = (int)(tile_d1 & 0xffffu);
  g1[5] = (int)(unsigned)(stride0 & 0xffffffffull);
  g1[6] = (int)(unsigned)((stride0 >> 32) & 0xffffull);
  g1[7] = 0;
  v4i z4 = {0, 0, 0, 0};
#if TDM_ARGS6
  v8i z8 = {0, 0, 0, 0, 0, 0, 0, 0};
  __builtin_amdgcn_tensor_load_to_lds(g0, g1, z4, z4, z8, 0);
#else
  __builtin_amdgcn_tensor_load_to_lds(g0, g1, z4, z4, 0);
#endif
}

// Load a 16x32 bf16 A-fragment (or symmetric NxK B-fragment) from LDS.
// ISA layout: lanes 0-15: row m=lane, K=0..7 & 16..23; lanes 16-31: K=8..15 & 24..31.
__device__ __forceinline__ v16bf ld_frag_bf16(const bf16_t* p, int stride, int lane) {
  const int m = lane & 15, kg = lane >> 4;
  const bf16_t* q = p + m * stride + 8 * kg;
  union { uint4 u[2]; v16bf f; } U;
  U.u[0] = *reinterpret_cast<const uint4*>(q);
  U.u[1] = *reinterpret_cast<const uint4*>(q + 16);
  return U.f;
}

// ---------------------------------------------------------------------------
// Prep 1: Y = bf16(X * rowmask[row]) ; X flat [4096 x 1024] fp32.
// ---------------------------------------------------------------------------
__global__ __launch_bounds__(256)
void cvt_mask_kernel(const float* __restrict__ X, const float* __restrict__ mask,
                     bf16_t* __restrict__ Y) {
  const size_t i0 = ((size_t)blockIdx.x * 256 + threadIdx.x) * 4;
  const float m = mask[i0 >> 10];
  const float4 x = *reinterpret_cast<const float4*>(X + i0);
  Y[i0 + 0] = (bf16_t)(x.x * m);
  Y[i0 + 1] = (bf16_t)(x.y * m);
  Y[i0 + 2] = (bf16_t)(x.z * m);
  Y[i0 + 3] = (bf16_t)(x.w * m);
}

// ---------------------------------------------------------------------------
// Prep 2: Wt[n][k] = bf16(W[k][n])  (1024x1024, LDS-tiled transpose)
// ---------------------------------------------------------------------------
__global__ __launch_bounds__(256)
void wtrans_kernel(const float* __restrict__ W, bf16_t* __restrict__ Wt) {
  __shared__ float tile[32][33];
  const int tx = threadIdx.x, ty = threadIdx.y;  // block (32,8)
  const int nb = blockIdx.x * 32, kb = blockIdx.y * 32;
  for (int j = ty; j < 32; j += 8)
    tile[j][tx] = W[(size_t)(kb + j) * 1024 + nb + tx];
  __syncthreads();
  for (int j = ty; j < 32; j += 8)
    Wt[(size_t)(nb + j) * 1024 + kb + tx] = (bf16_t)tile[tx][j];
}

// ---------------------------------------------------------------------------
// Projection GEMM:  Y = act(A @ Wt^T + bias),  A: [4096x1024] bf16 (pre-masked),
// Wt: [n][k] bf16.  Block: 256 thr (8 waves), tile 128x128, K staged 32/step
// via TDM into double-buffered LDS (rows padded to 80B by TDM pad_enable).
// ---------------------------------------------------------------------------
__global__ __launch_bounds__(256)
void proj_gemm_kernel(const bf16_t* __restrict__ A, const bf16_t* __restrict__ Wt,
                      const float* __restrict__ bias, void* __restrict__ Yout,
                      int leaky, int headOutBf16) {
  constexpr int AS = 40;  // padded LDS row stride (bf16 elems) = 80B
  __shared__ __align__(16) bf16_t Ab[2][128 * AS];
  __shared__ __align__(16) bf16_t Bb[2][128 * AS];

  const int tid = threadIdx.x, lane = tid & 31, wave = tid >> 5;
  const int wm = wave >> 2, wn = wave & 3;  // 2x4 wave grid
  const int m0 = blockIdx.x * 128, n0 = blockIdx.y * 128;

  v8f acc[4][2];
#pragma unroll
  for (int i = 0; i < 4; ++i)
#pragma unroll
    for (int j = 0; j < 2; ++j) acc[i][j] = v8f_zero();

  auto issue_tiles = [&](int buf, int k0) {
    // A tile: 128 rows (m) x 32 k ; B tile: 128 rows (n) x 32 k
    tdm_load_2d_bf16(A + (size_t)m0 * 1024 + k0, (unsigned)(uintptr_t)&Ab[buf][0],
                     1024u, 4096u, 1024ull, 32u, 128u, /*pad every 64B*/ 3u, /*+16B*/ 3u);
    tdm_load_2d_bf16(Wt + (size_t)n0 * 1024 + k0, (unsigned)(uintptr_t)&Bb[buf][0],
                     1024u, 1024u, 1024ull, 32u, 128u, 3u, 3u);
  };

  if (wave == 0) issue_tiles(0, 0);

  for (int it = 0; it < 32; ++it) {
    const int buf = it & 1;
    if (wave == 0) {
      if (it + 1 < 32) {
        issue_tiles(buf ^ 1, (it + 1) * 32);
        __builtin_amdgcn_s_wait_tensorcnt(2);  // current pair landed
      } else {
        __builtin_amdgcn_s_wait_tensorcnt(0);
      }
    }
    __syncthreads();  // tiles visible to all waves

    v16bf bfr[2];
#pragma unroll
    for (int nt = 0; nt < 2; ++nt)
      bfr[nt] = ld_frag_bf16(&Bb[buf][(wn * 32 + nt * 16) * AS], AS, lane);
#pragma unroll
    for (int mt = 0; mt < 4; ++mt) {
      v16bf afr = ld_frag_bf16(&Ab[buf][(wm * 64 + mt * 16) * AS], AS, lane);
#pragma unroll
      for (int nt = 0; nt < 2; ++nt)
        acc[mt][nt] = WMMA_BF16(afr, bfr[nt], acc[mt][nt]);
    }
    __syncthreads();  // done reading `buf`; TDM may overwrite it next round
  }

  // --- epilogue: bias, leaky, store ---
  const int nl = lane & 15, kg = lane >> 4;
#pragma unroll
  for (int mt = 0; mt < 4; ++mt)
#pragma unroll
    for (int nt = 0; nt < 2; ++nt) {
      const int Cc = n0 + wn * 32 + nt * 16 + nl;
      const float bv = bias[Cc];
#pragma unroll
      for (int r = 0; r < 8; ++r) {
        const int R = m0 + wm * 64 + mt * 16 + r + 8 * kg;
        float v = acc[mt][nt][r] + bv;
        if (leaky && v < 0.0f) v *= 0.1f;
        if (headOutBf16) {  // [B,H,S,64] bf16 for the attention kernel
          const int b = R >> 11, s = R & 2047, h = Cc >> 6, dd = Cc & 63;
          ((bf16_t*)Yout)[((size_t)((b << 4) + h) * 2048 + s) * 64 + dd] = (bf16_t)v;
        } else {            // final [B,S,D] fp32 output
          ((float*)Yout)[(size_t)R * 1024 + Cc] = v;
        }
      }
    }
}

// ---------------------------------------------------------------------------
// Chunked causal linear attention.  One block per (b,h); 128 thr = 4 waves.
// Chunk=64.  Wave w owns e-stripe [16w,16w+16): 4 fp32 state tiles in VGPRs.
// Output written directly in flat [4096 x 1024] bf16 (O-proj input layout).
// ---------------------------------------------------------------------------
__global__ __launch_bounds__(128)
void linear_attn_kernel(const bf16_t* __restrict__ qh, const bf16_t* __restrict__ kh,
                        const bf16_t* __restrict__ vh, bf16_t* __restrict__ outf) {
  constexpr int SLEN = 2048, NC = SLEN / 64;
  constexpr int LS = 72;  // padded bf16 LDS row stride: 144B
  __shared__ __align__(16) bf16_t Qb[64 * LS];   // [t][d]   (TDM)
  __shared__ __align__(16) bf16_t Kb[64 * LS];   // [t][d]   (TDM)
  __shared__ __align__(16) bf16_t Kbt[64 * LS];  // [d][t]
  __shared__ __align__(16) bf16_t Vtb[64 * LS];  // [e][t]
  __shared__ __align__(16) bf16_t Sb[64 * LS];   // [e][d]   state (bf16 view)
  __shared__ __align__(16) bf16_t Ab[64 * LS];   // [t][t']  masked attn weights
  __shared__ float den[64];
  __shared__ float zprev[64];

  const int tid = threadIdx.x, lane = tid & 31, wave = tid >> 5;
  const int bh = blockIdx.x, b = bh >> 4, h = bh & 15;
  const bf16_t* Qg = qh + (size_t)bh * SLEN * 64;
  const bf16_t* Kg = kh + (size_t)bh * SLEN * 64;
  const bf16_t* Vg = vh + (size_t)bh * SLEN * 64;
  bf16_t* Ob = outf + (size_t)b * 2048 * 1024 + h * 64;  // row stride 1024

  v8f st[4];
#pragma unroll
  for (int i = 0; i < 4; ++i) st[i] = v8f_zero();
  if (tid < 64) zprev[tid] = 0.0f;

  for (int c = 0; c < NC; ++c) {
    // ---- stage: TDM Q,K row tiles; VALU-stage transposed K,V; dump state ----
    if (wave == 0) {
      tdm_load_2d_bf16(Qg + (size_t)c * 64 * 64, (unsigned)(uintptr_t)Qb,
                       64u, 2048u, 64ull, 64u, 64u, /*pad every 128B*/ 4u, /*+16B*/ 3u);
      tdm_load_2d_bf16(Kg + (size_t)c * 64 * 64, (unsigned)(uintptr_t)Kb,
                       64u, 2048u, 64ull, 64u, 64u, 4u, 3u);
    }
    for (int idx = tid; idx < 4096; idx += 128) {
      const int t = idx >> 6, d = idx & 63;
      const size_t g = (size_t)(c * 64 + t) * 64 + d;
      Kbt[d * LS + t] = Kg[g];
      Vtb[d * LS + t] = Vg[g];
    }
    {
      const int n = lane & 15, kg = lane >> 4;
#pragma unroll
      for (int dt = 0; dt < 4; ++dt)
#pragma unroll
        for (int r = 0; r < 8; ++r)
          Sb[(wave * 16 + n) * LS + dt * 16 + r + 8 * kg] = (bf16_t)st[dt][r];
    }
    if (tid < 64) den[tid] = 0.0f;
    if (wave == 0) __builtin_amdgcn_s_wait_tensorcnt(0);
    __syncthreads();

    // ---- phase 1: inter = Q @ S_prev ; fp32 z-scan + denominator ----
    v8f inter[4];
#pragma unroll
    for (int i = 0; i < 4; ++i) {
      inter[i] = v8f_zero();
#pragma unroll
      for (int kb = 0; kb < 64; kb += 32) {
        v16bf a = ld_frag_bf16(Qb + i * 16 * LS + kb, LS, lane);
        v16bf bb = ld_frag_bf16(Sb + wave * 16 * LS + kb, LS, lane);
        inter[i] = WMMA_BF16(a, bb, inter[i]);
      }
    }
    if (tid < 64) {  // thread d: running z, scatter q_t[d]*z_t[d] into den[t]
      const int d = tid;
      float run = zprev[d];
      for (int t = 0; t < 64; ++t) {
        run += (float)Kbt[d * LS + t];
        atomicAdd(&den[t], (float)Qb[t * LS + d] * run);
      }
      zprev[d] = run;
    }
    __syncthreads();

    // ---- phase 2: masked Q @ K^T -> Ab (wave owns t'-stripe) ----
    {
      const int jw = wave, n = lane & 15, kg = lane >> 4;
#pragma unroll
      for (int i = 0; i < 4; ++i) {
        if (jw > i) {
#pragma unroll
          for (int r = 0; r < 8; ++r)
            Ab[(i * 16 + r + 8 * kg) * LS + jw * 16 + n] = (bf16_t)0.0f;
        } else {
          v8f s = v8f_zero();
#pragma unroll
          for (int kb = 0; kb < 64; kb += 32) {
            v16bf a = ld_frag_bf16(Qb + i * 16 * LS + kb, LS, lane);
            v16bf bb = ld_frag_bf16(Kb + jw * 16 * LS + kb, LS, lane);
            s = WMMA_BF16(a, bb, s);
          }
#pragma unroll
          for (int r = 0; r < 8; ++r) {
            const int tm = i * 16 + r + 8 * kg, tn = jw * 16 + n;
            Ab[tm * LS + tn] = (bf16_t)((tn <= tm) ? s[r] : 0.0f);
          }
        }
      }
    }
    __syncthreads();

    // ---- phase 3: num = inter + A@V, divide, store; state += K^T@V ----
    {
      const int n = lane & 15, kg = lane >> 4;
#pragma unroll
      for (int i = 0; i < 4; ++i) {
        v8f num = inter[i];
#pragma unroll
        for (int kb = 0; kb < 64; kb += 32) {
          v16bf a = ld_frag_bf16(Ab + i * 16 * LS + kb, LS, lane);
          v16bf bb = ld_frag_bf16(Vtb + wave * 16 * LS + kb, LS, lane);
          num = WMMA_BF16(a, bb, num);
        }
#pragma unroll
        for (int r = 0; r < 8; ++r) {
          const int tm = i * 16 + r + 8 * kg;
          Ob[(size_t)(c * 64 + tm) * 1024 + wave * 16 + n] = (bf16_t)(num[r] / den[tm]);
        }
      }
    }
#pragma unroll
    for (int dt = 0; dt < 4; ++dt)
#pragma unroll
      for (int kb = 0; kb < 64; kb += 32) {
        v16bf a = ld_frag_bf16(Kbt + dt * 16 * LS + kb, LS, lane);
        v16bf bb = ld_frag_bf16(Vtb + wave * 16 * LS + kb, LS, lane);
        st[dt] = WMMA_BF16(a, bb, st[dt]);
      }
    __syncthreads();
  }
}

// ---------------------------------------------------------------------------
extern "C" void kernel_launch(void* const* d_in, const int* in_sizes, int n_in,
                              void* d_out, int out_size, void* d_ws, size_t ws_size,
                              hipStream_t stream) {
  const float* q  = (const float*)d_in[0];
  const float* k  = (const float*)d_in[1];
  const float* v  = (const float*)d_in[2];
  const float* qm = (const float*)d_in[3];
  const float* km = (const float*)d_in[4];
  const float* vm = (const float*)d_in[5];
  const float* Wq = (const float*)d_in[6];
  const float* bq = (const float*)d_in[7];
  const float* Wk = (const float*)d_in[8];
  const float* bk = (const float*)d_in[9];
  const float* Wv = (const float*)d_in[10];
  const float* bv = (const float*)d_in[11];
  const float* Wo = (const float*)d_in[12];
  const float* bo = (const float*)d_in[13];
  float* out = (float*)d_out;

  const size_t NX = (size_t)4096 * 1024;  // activation-sized bf16 buffer
  const size_t NW = (size_t)1024 * 1024;  // weight-sized bf16 buffer
  bf16_t* Xq  = (bf16_t*)d_ws;
  bf16_t* Xk  = Xq + NX;
  bf16_t* Xv  = Xk + NX;
  bf16_t* Wtq = Xv + NX;
  bf16_t* Wtk = Wtq + NW;
  bf16_t* Wtv = Wtk + NW;
  bf16_t* Wto = Wtv + NW;
  bf16_t* qhb = Wto + NW;
  bf16_t* khb = qhb + NX;
  bf16_t* vhb = khb + NX;
  bf16_t* aof = vhb + NX;

  // prep: mask+cast activations; transpose+cast weights
  cvt_mask_kernel<<<dim3(4096), 256, 0, stream>>>(q, qm, Xq);
  cvt_mask_kernel<<<dim3(4096), 256, 0, stream>>>(k, km, Xk);
  cvt_mask_kernel<<<dim3(4096), 256, 0, stream>>>(v, vm, Xv);
  wtrans_kernel<<<dim3(32, 32), dim3(32, 8), 0, stream>>>(Wq, Wtq);
  wtrans_kernel<<<dim3(32, 32), dim3(32, 8), 0, stream>>>(Wk, Wtk);
  wtrans_kernel<<<dim3(32, 32), dim3(32, 8), 0, stream>>>(Wv, Wtv);
  wtrans_kernel<<<dim3(32, 32), dim3(32, 8), 0, stream>>>(Wo, Wto);

  dim3 pgrid(4096 / 128, 1024 / 128);  // 32 x 8 blocks
  proj_gemm_kernel<<<pgrid, 256, 0, stream>>>(Xq, Wtq, bq, qhb, 1, 1);
  proj_gemm_kernel<<<pgrid, 256, 0, stream>>>(Xk, Wtk, bk, khb, 1, 1);
  proj_gemm_kernel<<<pgrid, 256, 0, stream>>>(Xv, Wtv, bv, vhb, 0, 1);
  linear_attn_kernel<<<dim3(32), 128, 0, stream>>>(qhb, khb, vhb, aof);
  proj_gemm_kernel<<<pgrid, 256, 0, stream>>>(aof, Wto, bo, (void*)out, 0, 0);
}
